// Attention_73031623901451
// MI455X (gfx1250) — compile-verified, matrix operand order checked by reading, not run on hardware
//
#include <hip/hip_runtime.h>

#define BB 8
#define CC 256
#define NN 2048

typedef __attribute__((ext_vector_type(16))) _Float16 v16h;
typedef __attribute__((ext_vector_type(8)))  float    v8f;

union V16H { uint4 u4[2]; v16h h; };
union F8   { v8f v; float f[8]; };
union H8   { _Float16 h[8]; uint4 u; };

// Register-blocked MTxNT macro-tile (16*MT x 16*NT output) per wave.
// A : row-major [M][K] (lda halves); Bt: column list Bt[n][k] (ldb halves).
// K multiple of 32. Operand layout matches wave32 WMMA f16 A/B layout:
// per lane two contiguous 8-half runs at k + g*8 and k + 16 + g*8.
template <int MT, int NT>
__device__ inline void gemm_tile(const _Float16* __restrict__ A, int lda,
                                 const _Float16* __restrict__ Bt, int ldb,
                                 int K, int tm, int tn, int lane,
                                 v8f acc[MT][NT]) {
  const int l15 = lane & 15;
  const int g   = (lane >> 4) & 1;
  const int off0 = g * 8;
  const int off1 = 16 + g * 8;
  const _Float16* ap[MT];
  const _Float16* bp[NT];
#pragma unroll
  for (int mi = 0; mi < MT; ++mi) ap[mi] = A + (size_t)(tm + 16 * mi + l15) * lda;
#pragma unroll
  for (int ni = 0; ni < NT; ++ni) bp[ni] = Bt + (size_t)(tn + 16 * ni + l15) * ldb;
  v8f z = {};
#pragma unroll
  for (int mi = 0; mi < MT; ++mi)
#pragma unroll
    for (int ni = 0; ni < NT; ++ni) acc[mi][ni] = z;

  for (int kk = 0; kk < K; kk += 32) {
    V16H va[MT], vb[NT];
#pragma unroll
    for (int mi = 0; mi < MT; ++mi) {
      va[mi].u4[0] = *(const uint4*)(ap[mi] + kk + off0);
      va[mi].u4[1] = *(const uint4*)(ap[mi] + kk + off1);
    }
#pragma unroll
    for (int ni = 0; ni < NT; ++ni) {
      vb[ni].u4[0] = *(const uint4*)(bp[ni] + kk + off0);
      vb[ni].u4[1] = *(const uint4*)(bp[ni] + kk + off1);
    }
#pragma unroll
    for (int mi = 0; mi < MT; ++mi)
#pragma unroll
      for (int ni = 0; ni < NT; ++ni)
        acc[mi][ni] = __builtin_amdgcn_wmma_f32_16x16x32_f16(
            false, va[mi].h, false, vb[ni].h, (short)0, acc[mi][ni], false, false);
  }
}

// ---- Kernel 1: x[b][c][n] fp32 -> xt[b][n][c] f16 (tiled LDS transpose) ----
__global__ void k_transpose_x(const float* __restrict__ x, _Float16* __restrict__ xt) {
  __shared__ _Float16 tile[32][33];
  const int b  = blockIdx.z;
  const int ct = blockIdx.y * 32;
  const int nt = blockIdx.x * 32;
  const int tx = threadIdx.x & 31;
  const int ty = threadIdx.x >> 5; // 0..7
  const float* xp = x + ((size_t)b * CC + ct) * NN + nt;
#pragma unroll
  for (int i = 0; i < 4; ++i) {
    int c = ty + 8 * i;
    tile[c][tx] = (_Float16)xp[(size_t)c * NN + tx];
  }
  __syncthreads();
  _Float16* xo = xt + ((size_t)b * NN + nt) * CC + ct;
#pragma unroll
  for (int i = 0; i < 4; ++i) {
    int n = ty + 8 * i;
    xo[(size_t)n * CC + tx] = tile[tx][n];
  }
}

// ---- Kernel 2: W fp32 -> f16 (3 matrices) ----
__global__ void k_cvt_w(const float* __restrict__ wq, const float* __restrict__ wk,
                        const float* __restrict__ wv,
                        _Float16* __restrict__ oq, _Float16* __restrict__ ok,
                        _Float16* __restrict__ ov) {
  const int which = blockIdx.y;
  const float* src = (which == 0) ? wq : (which == 1) ? wk : wv;
  _Float16*    dst = (which == 0) ? oq : (which == 1) ? ok : ov;
  const int i = blockIdx.x * 256 + threadIdx.x;
  dst[i] = (_Float16)src[i];
}

// ---- Kernel 3: sx[b*C+c] = sum_n x[b][c][n] (one wave per row) ----
__global__ void k_rowsum_x(const float* __restrict__ x, float* __restrict__ sx) {
  const int wave = (int)((blockIdx.x * blockDim.x + threadIdx.x) >> 5);
  const int lane = threadIdx.x & 31;
  const float* row = x + (size_t)wave * NN;
  float s = 0.f;
  for (int i = lane; i < NN; i += 32) s += row[i];
#pragma unroll
  for (int o = 16; o; o >>= 1) s += __shfl_xor(s, o, 32);
  if (lane == 0) sx[wave] = s;
}

// ---- Kernel 4: invS = 1 / sum_{b,c} (Wq[c,:]*sx[b,:]+N*bq[c]) * (Wk[c,:]*sx[b,:]+N*bk[c])
__global__ void k_scalar_s(const float* __restrict__ Wq, const float* __restrict__ bq,
                           const float* __restrict__ Wk, const float* __restrict__ bk,
                           const float* __restrict__ sx, float* __restrict__ invS) {
  __shared__ float red[256];
  float part = 0.f;
  for (int p = threadIdx.x; p < BB * CC; p += 256) {
    const int b = p / CC, c = p % CC;
    const float* sxb = sx + b * CC;
    const float* wq = Wq + (size_t)c * CC;
    const float* wk = Wk + (size_t)c * CC;
    float aq = 0.f, ak = 0.f;
    for (int j = 0; j < CC; ++j) { float s = sxb[j]; aq += wq[j] * s; ak += wk[j] * s; }
    aq += (float)NN * bq[c];
    ak += (float)NN * bk[c];
    part += aq * ak;
  }
  red[threadIdx.x] = part;
  __syncthreads();
  for (int o = 128; o; o >>= 1) {
    if (threadIdx.x < o) red[threadIdx.x] += red[threadIdx.x + o];
    __syncthreads();
  }
  if (threadIdx.x == 0) invS[0] = 1.0f / red[0];
}

// ---- Kernel 5: q/k/v = W @ x + b.  q,v -> [b][c][n] f16 ; k -> kt[b][n][c] f16
// 64x64 per wave (4x4 WMMA macro-tile): 16 b128 loads : 16 wmma per k-step.
__global__ void __launch_bounds__(256)
k_qkv(const _Float16* __restrict__ xt,
      const _Float16* __restrict__ wq, const _Float16* __restrict__ wk,
      const _Float16* __restrict__ wv,
      const float* __restrict__ bq, const float* __restrict__ bk,
      const float* __restrict__ bv,
      _Float16* __restrict__ qh, _Float16* __restrict__ kt,
      _Float16* __restrict__ vh) {
  const int b     = blockIdx.z & 7;
  const int which = blockIdx.z >> 3; // 0=q 1=k 2=v
  const _Float16* W    = (which == 0) ? wq : (which == 1) ? wk : wv;
  const float*    bias = (which == 0) ? bq : (which == 1) ? bk : bv;
  const int wave = threadIdx.x >> 5;
  const int lane = threadIdx.x & 31;
  const int tile = blockIdx.x * 8 + wave;    // 0..127
  const int tm = (tile & 3) * 64;            // C/64  = 4 tiles over out-channels
  const int tn = (tile >> 2) * 64;           // N/64 = 32 tiles over positions
  const _Float16* Bt = xt + (size_t)b * NN * CC;   // Bt[n][c], ld = C
  v8f acc[4][4];
  gemm_tile<4, 4>(W, CC, Bt, CC, CC, tm, tn, lane, acc);
  const int l15 = lane & 15, g = (lane >> 4) & 1;
#pragma unroll
  for (int mi = 0; mi < 4; ++mi) {
#pragma unroll
    for (int ni = 0; ni < 4; ++ni) {
      F8 f; f.v = acc[mi][ni];
      const int mb = tm + 16 * mi;
      const int n  = tn + 16 * ni + l15;
      if (which == 1) {
        H8 pk;
#pragma unroll
        for (int r = 0; r < 8; ++r) { int m = mb + r + 8 * g; pk.h[r] = (_Float16)(f.f[r] + bias[m]); }
        *(uint4*)(kt + ((size_t)b * NN + n) * CC + mb + 8 * g) = pk.u;
      } else {
        _Float16* out = ((which == 0) ? qh : vh) + (size_t)b * CC * NN;
#pragma unroll
        for (int r = 0; r < 8; ++r) {
          int m = mb + r + 8 * g;
          out[(size_t)m * NN + n] = (_Float16)(f.f[r] + bias[m]);
        }
      }
    }
  }
}

// ---- Kernel 6: M2[b] = v[b] @ q[b]^T  (C x C), K = N ----
// Small output per batch: keep 32x32 tiles (2x2) for parallelism.
__global__ void __launch_bounds__(256)
k_m2(const _Float16* __restrict__ vh, const _Float16* __restrict__ qh,
     _Float16* __restrict__ m2h) {
  const int b = blockIdx.z;
  const int wave = threadIdx.x >> 5;
  const int lane = threadIdx.x & 31;
  const int tile = blockIdx.x * 8 + wave;  // 0..63
  const int tm = (tile & 7) * 32;
  const int tn = (tile >> 3) * 32;
  const _Float16* A  = vh + (size_t)b * CC * NN;  // [c][n], ld = N
  const _Float16* Bt = qh + (size_t)b * CC * NN;  // col c' -> q[c'][:], ld = N
  v8f acc[2][2];
  gemm_tile<2, 2>(A, NN, Bt, NN, NN, tm, tn, lane, acc);
  const int l15 = lane & 15, g = (lane >> 4) & 1;
  _Float16* out = m2h + (size_t)b * CC * CC;
#pragma unroll
  for (int mi = 0; mi < 2; ++mi) {
#pragma unroll
    for (int ni = 0; ni < 2; ++ni) {
      F8 f; f.v = acc[mi][ni];
      const int mb = tm + 16 * mi;
      const int cn = tn + 16 * ni + l15;
#pragma unroll
      for (int r = 0; r < 8; ++r) {
        int m = mb + r + 8 * g;
        out[(size_t)m * CC + cn] = (_Float16)f.f[r];
      }
    }
  }
}

// ---- Kernel 7: attn[b] = (M2[b] @ k[b]) * invS  -> fp32 out [b][c][n] ----
__global__ void __launch_bounds__(256)
k_attn(const _Float16* __restrict__ m2h, const _Float16* __restrict__ kt,
       const float* __restrict__ invS, float* __restrict__ out) {
  const int b = blockIdx.z;
  const int wave = threadIdx.x >> 5;
  const int lane = threadIdx.x & 31;
  const int tile = blockIdx.x * 8 + wave;  // 0..127
  const int tm = (tile & 3) * 64;
  const int tn = (tile >> 2) * 64;
  const _Float16* A  = m2h + (size_t)b * CC * CC;  // [c][c'], ld = C
  const _Float16* Bt = kt  + (size_t)b * NN * CC;  // Bt[n][c'], ld = C
  v8f acc[4][4];
  gemm_tile<4, 4>(A, CC, Bt, CC, CC, tm, tn, lane, acc);
  const float s = invS[0];
  const int l15 = lane & 15, g = (lane >> 4) & 1;
  float* ob = out + (size_t)b * CC * NN;
#pragma unroll
  for (int mi = 0; mi < 4; ++mi) {
#pragma unroll
    for (int ni = 0; ni < 4; ++ni) {
      F8 f; f.v = acc[mi][ni];
      const int mb = tm + 16 * mi;
      const int n  = tn + 16 * ni + l15;
#pragma unroll
      for (int r = 0; r < 8; ++r) {
        int m = mb + r + 8 * g;
        ob[(size_t)m * NN + n] = f.f[r] * s;
      }
    }
  }
}

extern "C" void kernel_launch(void* const* d_in, const int* in_sizes, int n_in,
                              void* d_out, int out_size, void* d_ws, size_t ws_size,
                              hipStream_t stream) {
  const float* x  = (const float*)d_in[0];
  const float* Wq = (const float*)d_in[1];
  const float* bq = (const float*)d_in[2];
  const float* Wk = (const float*)d_in[3];
  const float* bk = (const float*)d_in[4];
  const float* Wv = (const float*)d_in[5];
  const float* bv = (const float*)d_in[6];
  float* out = (float*)d_out;

  char* ws = (char*)d_ws;
  size_t off = 0;
  _Float16* xt  = (_Float16*)(ws + off); off += (size_t)BB * NN * CC * 2;   // 8 MB
  _Float16* whq = (_Float16*)(ws + off); off += (size_t)CC * CC * 2;
  _Float16* whk = (_Float16*)(ws + off); off += (size_t)CC * CC * 2;
  _Float16* whv = (_Float16*)(ws + off); off += (size_t)CC * CC * 2;
  _Float16* qh  = (_Float16*)(ws + off); off += (size_t)BB * CC * NN * 2;   // 8 MB
  _Float16* kt  = (_Float16*)(ws + off); off += (size_t)BB * NN * CC * 2;   // 8 MB
  _Float16* vh  = (_Float16*)(ws + off); off += (size_t)BB * CC * NN * 2;   // 8 MB
  _Float16* m2h = (_Float16*)(ws + off); off += (size_t)BB * CC * CC * 2;   // 1 MB
  float*    sx  = (float*)(ws + off);    off += (size_t)BB * CC * 4;
  float*    invS= (float*)(ws + off);    off += 256;

  // 1. transpose+convert x
  k_transpose_x<<<dim3(NN / 32, CC / 32, BB), 256, 0, stream>>>(x, xt);
  // 2. convert weights
  k_cvt_w<<<dim3(CC * CC / 256, 3), 256, 0, stream>>>(Wq, Wk, Wv, whq, whk, whv);
  // 3. row sums of x
  k_rowsum_x<<<dim3(BB * CC / 8), 256, 0, stream>>>(x, sx);
  // 4. global scalar S (from fp32 weights for accuracy)
  k_scalar_s<<<dim3(1), 256, 0, stream>>>(Wq, bq, Wk, bk, sx, invS);
  // 5. q/k/v GEMMs (WMMA), z = batch*3
  k_qkv<<<dim3(16, 1, BB * 3), 256, 0, stream>>>(xt, whq, whk, whv, bq, bk, bv, qh, kt, vh);
  // 6. M2 = v q^T per batch (WMMA)
  k_m2<<<dim3(8, 1, BB), 256, 0, stream>>>(vh, qh, m2h);
  // 7. attn = (M2 k) / S per batch (WMMA)
  k_attn<<<dim3(16, 1, BB), 256, 0, stream>>>(m2h, kt, invS, out);
  (void)in_sizes; (void)n_in; (void)out_size; (void)ws_size;
}
